// IndRNN_1460288880806
// MI455X (gfx1250) — matrix operand order repaired and checked by reading, not compiled, not run on hardware
//

#include <hip/hip_runtime.h>
#include <hip/hip_bf16.h>

// IndRNN (2 layers), SEQ=4096, BATCH=32, D=512 on gfx1250 (MI455X).
// Per layer: proj = x @ W^T + b  (f16 WMMA, fp32 accum), then in-place
// sequential scan h = relu(p + w*h). Entire pipeline is staged in d_out.

typedef __attribute__((ext_vector_type(16))) _Float16 v16h;
typedef __attribute__((ext_vector_type(8)))  float    v8f;
typedef __attribute__((ext_vector_type(2)))  __fp16   pkh2; // cvt_pkrtz return type

#define TSEQ  4096
#define NBAT  32
#define DIM   512
#define SROW  (NBAT * DIM)     /* 16384 elements per time step  */
#define MROWS (TSEQ * NBAT)    /* 131072 GEMM rows              */

__device__ __forceinline__ void pk2(v16h& d, int i, float a, float b) {
    pkh2 t = __builtin_amdgcn_cvt_pkrtz(a, b);   // packed f32->f16 convert
    d[i] = (_Float16)t[0]; d[i + 1] = (_Float16)t[1];
}

// ---------------------------------------------------------------------------
// GEMM: P[m, h] = sum_i X[m, i] * W[h, i] + bias[h]
// Grid: MROWS/32 blocks x 256 threads (8 waves).
// Wave `nw` owns N-cols [nw*64, nw*64+64); block owns M-rows [bx*32, bx*32+32).
// Safe for P == X (in-place): each row is read only by its own workgroup and
// all K-loop loads complete before the __syncthreads() that precedes stores.
// ---------------------------------------------------------------------------
__global__ __launch_bounds__(256, 1) void indrnn_gemm(
    const float* X, const float* W, const float* bias, float* P)
{
    const int lane = threadIdx.x & 31;
    const int wave = threadIdx.x >> 5;
    const int lh   = lane & 15;       // 0..15
    const int hi   = lane >> 4;       // 0 or 1 (half-wave select)
    const int n0   = wave * 64;
    const int m0   = blockIdx.x * 32;

    v8f acc[2][4];
#pragma unroll
    for (int nt = 0; nt < 4; ++nt) {
        float b = bias[n0 + nt * 16 + lh];
#pragma unroll
        for (int ms = 0; ms < 2; ++ms)
#pragma unroll
            for (int j = 0; j < 8; ++j) acc[ms][nt][j] = b;
    }

    const float* xr0 = X + (size_t)(m0 + lh) * DIM;
    const float* xr1 = X + (size_t)(m0 + 16 + lh) * DIM;

    for (int k0 = 0; k0 < DIM; k0 += 32) {
        // B fragments: column h = n0+nt*16+lh, K = k0 + hi*16 + (0..15)
        v16h B[4];
#pragma unroll
        for (int nt = 0; nt < 4; ++nt) {
            const float4* wr =
                (const float4*)(W + (size_t)(n0 + nt * 16 + lh) * DIM + k0 + hi * 16);
            float4 b0 = wr[0], b1 = wr[1], b2 = wr[2], b3 = wr[3];
            pk2(B[nt], 0,  b0.x, b0.y); pk2(B[nt], 2,  b0.z, b0.w);
            pk2(B[nt], 4,  b1.x, b1.y); pk2(B[nt], 6,  b1.z, b1.w);
            pk2(B[nt], 8,  b2.x, b2.y); pk2(B[nt], 10, b2.z, b2.w);
            pk2(B[nt], 12, b3.x, b3.y); pk2(B[nt], 14, b3.z, b3.w);
        }
        // A fragments: row m = strip_base + lh,
        // K = k0 + hi*16 + (0..15)
        v16h A[2];
#pragma unroll
        for (int ms = 0; ms < 2; ++ms) {
            const float* xr = ms ? xr1 : xr0;
            const int aoff = k0 + hi * 16;
            float4 a0 = *(const float4*)(xr + aoff);
            float4 a1 = *(const float4*)(xr + aoff + 4);
            float4 a2 = *(const float4*)(xr + aoff + 8);
            float4 a3 = *(const float4*)(xr + aoff + 12);
            pk2(A[ms], 0,  a0.x, a0.y); pk2(A[ms], 2,  a0.z, a0.w);
            pk2(A[ms], 4,  a1.x, a1.y); pk2(A[ms], 6,  a1.z, a1.w);
            pk2(A[ms], 8,  a2.x, a2.y); pk2(A[ms], 10, a2.z, a2.w);
            pk2(A[ms], 12, a3.x, a3.y); pk2(A[ms], 14, a3.z, a3.w);
        }
#pragma unroll
        for (int ms = 0; ms < 2; ++ms)
#pragma unroll
            for (int nt = 0; nt < 4; ++nt)
                acc[ms][nt] = __builtin_amdgcn_wmma_f32_16x16x32_f16(
                    false, A[ms], false, B[nt], (short)0, acc[ms][nt],
                    false, false);
    }

    __syncthreads();  // in-place hazard fence: all WG loads done before stores

#pragma unroll
    for (int ms = 0; ms < 2; ++ms)
#pragma unroll
        for (int nt = 0; nt < 4; ++nt) {
            const int col = n0 + nt * 16 + lh;
#pragma unroll
            for (int j = 0; j < 8; ++j) {
                const int row = m0 + ms * 16 + hi * 8 + j;
                P[(size_t)row * DIM + col] = acc[ms][nt][j];
            }
        }
}

// ---------------------------------------------------------------------------
// In-place IndRNN scan: buf[t] <- h_t = relu(buf[t] + w .* h_{t-1}).
// One thread per (b, d) element; 16-deep unroll => 16 independent outstanding
// loads per thread to hide HBM latency at 64-workgroup occupancy.
// ---------------------------------------------------------------------------
__global__ __launch_bounds__(256, 1) void indrnn_scan(
    float* buf, const float* w_hh, float* hn)
{
    const int idx = blockIdx.x * 256 + threadIdx.x;   // 0..16383
    const int d   = idx & (DIM - 1);
    const float w = w_hh[d];
    float h = 0.0f;
    float* p = buf + idx;

    for (int t = 0; t < TSEQ; t += 16) {
        float v[16];
#pragma unroll
        for (int j = 0; j < 16; ++j) v[j] = p[(size_t)(t + j) * SROW];
#pragma unroll
        for (int j = 0; j < 16; ++j) {
            h = fmaxf(fmaf(w, h, v[j]), 0.0f);
            v[j] = h;
        }
#pragma unroll
        for (int j = 0; j < 16; ++j) p[(size_t)(t + j) * SROW] = v[j];
    }
    hn[idx] = h;
}

// ---------------------------------------------------------------------------
extern "C" void kernel_launch(void* const* d_in, const int* in_sizes, int n_in,
                              void* d_out, int out_size, void* d_ws, size_t ws_size,
                              hipStream_t stream) {
    (void)in_sizes; (void)n_in; (void)out_size; (void)d_ws; (void)ws_size;

    const float* x    = (const float*)d_in[0];   // (4096, 32, 512)
    const float* W_ih = (const float*)d_in[1];   // (2, 512, 512)
    const float* w_hh = (const float*)d_in[2];   // (2, 512)
    const float* b_ih = (const float*)d_in[3];   // (2, 512)

    float* out = (float*)d_out;                  // (4096, 32, 512)
    float* hn  = out + (size_t)MROWS * DIM;      // (2, 32, 512)

    const dim3 gGemm(MROWS / 32), bGemm(256);
    const dim3 gScan(SROW / 256), bScan(256);

    // Layer 1: proj1 -> d_out, scan in place, h_n[0]
    indrnn_gemm<<<gGemm, bGemm, 0, stream>>>(x, W_ih, b_ih, out);
    indrnn_scan<<<gScan, bScan, 0, stream>>>(out, w_hh, hn);

    // Layer 2: proj2 computed in place from ys1 (row-wise in-place safe),
    // scan in place -> final output, h_n[1]
    indrnn_gemm<<<gGemm, bGemm, 0, stream>>>(out, W_ih + DIM * DIM,
                                             b_ih + DIM, out);
    indrnn_scan<<<gScan, bScan, 0, stream>>>(out, w_hh + DIM, hn + SROW);
}
